// HeteroForecastSageConv_85822036509302
// MI455X (gfx1250) — compile-verified, matrix operand order checked by reading, not run on hardware
//
#include <hip/hip_runtime.h>
#include <stddef.h>
#include <stdint.h>

// ---------------------------------------------------------------------------
// HeteroForecastSageConv on MI455X (gfx1250, wave32, WMMA bf16)
//
// Pipeline (all on `stream`):
//   P0  prep_weights : transpose/convert weights to bf16 N-major; build the
//                      fused K=512 weight [0.5(W_self+W_ct_r)+I ; .25 W_s2d ;
//                      .25 W_d2s ; .5 W_ct_l] and fused bias.
//   P1  conv f32->bf16 of x_target / x_context
//   P2  zero agg region of A_big + degree counters
//   K1  x_t = relu(x_target @ Wp_t + bp_t)   -> A_big[:,0:128]   (bf16)
//   K2  x_c = relu(x_context @ Wp_c + bp_c)  -> xc_h             (bf16)
//   E1  tt edges: gather bf16 rows, GLOBAL_ATOMIC_PK_ADD_BF16 into
//       A_big[:,128:256] (s2d) and A_big[:,256:384] (d2s) + f32 degree atomics
//   E2  ct edges: same into A_big[:,384:512]
//   N   normalize agg columns by 1/max(deg,1)
//   G1  z = relu(A_big(100000x512) @ Wcomb + b_comb)  -> zr (bf16)
//   G2  out = zr @ W_out + b_out                    -> d_out (f32)
// ---------------------------------------------------------------------------

typedef unsigned short u16;
typedef unsigned int   u32;

typedef __bf16 v16bf __attribute__((ext_vector_type(16)));
typedef float  v8f   __attribute__((ext_vector_type(8)));

union Frag16 { v16bf v; uint4 q[2]; };

static constexpr int NT  = 100000;
static constexpr int NC  = 20000;
static constexpr int ETT = 600000;
static constexpr int ECT = 400000;
static constexpr int H   = 128;

#if defined(__HIP_DEVICE_COMPILE__) && __has_builtin(__builtin_amdgcn_global_load_async_to_lds_b128)
#define HAVE_ASYNC_LDS 1
// Builtin signature (from clang diagnostic): param0 = int4 in device (AS1)
// address space; param1 = LDS-side pointer; then imm offset + cpol.
typedef __attribute__((__vector_size__(4 * sizeof(int)))) int v4i_t;
typedef __attribute__((address_space(1))) v4i_t GV4;
typedef __attribute__((address_space(3))) v4i_t LV4;
#else
#define HAVE_ASYNC_LDS 0
#endif

__device__ __forceinline__ void wait_asynccnt0() {
#if defined(__HIP_DEVICE_COMPILE__)
#if __has_builtin(__builtin_amdgcn_s_wait_asynccnt)
    __builtin_amdgcn_s_wait_asynccnt(0);
#else
    asm volatile("s_wait_asynccnt 0x0" ::: "memory");
#endif
#endif
}

__device__ __forceinline__ u16 f2bf(float f) {
    u32 u = __float_as_uint(f);
    u32 r = u + 0x7FFFu + ((u >> 16) & 1u);   // round-to-nearest-even
    return (u16)(r >> 16);
}
__device__ __forceinline__ float bf2f(u16 h) {
    return __uint_as_float(((u32)h) << 16);
}

// Packed bf16 atomic add (2 x bf16 per op) -- CDNA5 GLOBAL_ATOMIC_PK_ADD_BF16.
__device__ __forceinline__ void pk_add_bf16(u16* p, u32 packed) {
    asm volatile("global_atomic_pk_add_bf16 %0, %1, off"
                 :: "v"(p), "v"(packed) : "memory");
}

// ---------------------------------------------------------------------------
// P0: weight prep (transpose to N-major bf16, build fused weight/bias)
// ---------------------------------------------------------------------------
__global__ void prep_weights(const float* __restrict__ Wp_t,
                             const float* __restrict__ Wp_c,
                             const float* __restrict__ W_self,
                             const float* __restrict__ W_ct_r,
                             const float* __restrict__ W_s2d,
                             const float* __restrict__ W_d2s,
                             const float* __restrict__ W_ct_l,
                             const float* __restrict__ W_out,
                             const float* __restrict__ b_self,
                             const float* __restrict__ b_s2d,
                             const float* __restrict__ b_d2s,
                             const float* __restrict__ b_ct_l,
                             u16* __restrict__ WpT_t, u16* __restrict__ WpT_c,
                             u16* __restrict__ WcombT, u16* __restrict__ WoutT,
                             float* __restrict__ b_comb) {
    const int stride = gridDim.x * blockDim.x;
    const int g = blockIdx.x * blockDim.x + threadIdx.x;
    // WpT_t / WpT_c / WoutT: [n][k] = W[k][n]
    for (int i = g; i < H * H; i += stride) {
        int n = i >> 7, k = i & 127;
        WpT_t[i] = f2bf(Wp_t[k * H + n]);
        WpT_c[i] = f2bf(Wp_c[k * H + n]);
        WoutT[i] = f2bf(W_out[k * H + n]);
    }
    // WcombT: [n][k] over K=512
    for (int i = g; i < H * 512; i += stride) {
        int n = i >> 9, k = i & 511;
        float w;
        if (k < 128) {
            w = 0.5f * (W_self[k * H + n] + W_ct_r[k * H + n]) + (k == n ? 1.0f : 0.0f);
        } else if (k < 256) {
            w = 0.25f * W_s2d[(k - 128) * H + n];
        } else if (k < 384) {
            w = 0.25f * W_d2s[(k - 256) * H + n];
        } else {
            w = 0.5f * W_ct_l[(k - 384) * H + n];
        }
        WcombT[n * 512 + k] = f2bf(w);
    }
    for (int i = g; i < H; i += stride)
        b_comb[i] = 0.5f * (b_self[i] + b_ct_l[i]) + 0.25f * (b_s2d[i] + b_d2s[i]);
}

// ---------------------------------------------------------------------------
// P1: f32 -> bf16 convert
// ---------------------------------------------------------------------------
__global__ void conv_f32_bf16(const float* __restrict__ in, u16* __restrict__ out, int n) {
    int stride = gridDim.x * blockDim.x;
    for (int i = blockIdx.x * blockDim.x + threadIdx.x; i < n; i += stride)
        out[i] = f2bf(in[i]);
}

// ---------------------------------------------------------------------------
// P2: zero agg columns [128..512) of A_big (bf16) and counters
// ---------------------------------------------------------------------------
__global__ void zero_agg(u16* __restrict__ A_big) {
    // one uint4 (8 bf16) per thread: NT rows * 48 uint4 per row
    int stride = gridDim.x * blockDim.x;
    const uint4 z = {0u, 0u, 0u, 0u};
    for (int i = blockIdx.x * blockDim.x + threadIdx.x; i < NT * 48; i += stride) {
        int row = i / 48, c = i % 48;
        *(uint4*)(A_big + (size_t)row * 512 + 128 + c * 8) = z;
    }
}

__global__ void zero_f32(float* __restrict__ p, int n) {
    int stride = gridDim.x * blockDim.x;
    for (int i = blockIdx.x * blockDim.x + threadIdx.x; i < n; i += stride)
        p[i] = 0.0f;
}

// ---------------------------------------------------------------------------
// GEMM: C(MxN=128) = op(A(MxK,bf16) @ B(KxN,bf16 stored N-major) + bias)
// 256 threads = 8 waves; wave w -> rows [16w,16w+16), all 128 cols.
// B k-panel (32x128 bf16 = 8 KB) staged in LDS each k-step via
// GLOBAL_LOAD_ASYNC_TO_LDS_B128 (ASYNCcnt) when available, else
// global_load_b128 + ds_store_b128.
//
// 16-bit WMMA operand layout (ISA 7.12.2): lane l<16 holds (row=l) K chunks
// [0..7] and [16..23]; lane l>=16 holds K chunks [8..15] and [24..31].
// B is symmetric with N striped across lanes, so B from N-major memory uses
// identical addressing with row := n.
// ---------------------------------------------------------------------------
template <bool RELU, bool OUT_F32>
__global__ __launch_bounds__(256) void gemm_bf16_n128(
        const u16* __restrict__ A, int lda,
        const u16* __restrict__ BT, int K,
        const float* __restrict__ bias,
        void* __restrict__ outp, int ldo, int M) {
    __shared__ u16 ldsB[128 * 32];   // 8 KB: B panel for current k-step

    const int tid  = threadIdx.x;
    const int lane = tid & 31;
    const int wave = tid >> 5;
    const int l15  = lane & 15;
    const int half = lane >> 4;              // 0 or 1

    const int mTiles  = M >> 4;
    const int rowTile = blockIdx.x * 8 + wave;
    const bool active = rowTile < mTiles;
    const int rowBase = active ? (rowTile << 4) : 0;   // keep EXEC all-ones

    v8f acc[8] = {};

    const u16* aRow = A + (size_t)(rowBase + l15) * lda + half * 8;

    // staging indices for this thread: two b128 transfers per k-step
    const int li0 = tid, li1 = tid + 256;
    const int r0 = li0 >> 2, ch0 = li0 & 3;
    const int r1 = li1 >> 2, ch1 = li1 & 3;

    for (int k0 = 0; k0 < K; k0 += 32) {
#if HAVE_ASYNC_LDS
        __builtin_amdgcn_global_load_async_to_lds_b128(
            (GV4*)(BT + (size_t)r0 * K + k0 + ch0 * 8),
            (LV4*)&ldsB[r0 * 32 + ch0 * 8], 0, 0);
        __builtin_amdgcn_global_load_async_to_lds_b128(
            (GV4*)(BT + (size_t)r1 * K + k0 + ch1 * 8),
            (LV4*)&ldsB[r1 * 32 + ch1 * 8], 0, 0);
        wait_asynccnt0();
#else
        *(uint4*)&ldsB[r0 * 32 + ch0 * 8] = *(const uint4*)&BT[(size_t)r0 * K + k0 + ch0 * 8];
        *(uint4*)&ldsB[r1 * 32 + ch1 * 8] = *(const uint4*)&BT[(size_t)r1 * K + k0 + ch1 * 8];
#endif
        __syncthreads();

        Frag16 af;
        af.q[0] = *(const uint4*)(aRow + k0);
        af.q[1] = *(const uint4*)(aRow + k0 + 16);

        // preload ALL 8 B fragments, then issue 8 back-to-back WMMAs
        Frag16 bf[8];
        #pragma unroll
        for (int nt = 0; nt < 8; ++nt) {
            const u16* bp = &ldsB[(nt * 16 + l15) * 32 + half * 8];
            bf[nt].q[0] = *(const uint4*)bp;          // ds_load_b128
            bf[nt].q[1] = *(const uint4*)(bp + 16);   // ds_load_b128
        }
        #pragma unroll
        for (int nt = 0; nt < 8; ++nt) {
            acc[nt] = __builtin_amdgcn_wmma_f32_16x16x32_bf16(
                false, af.v, false, bf[nt].v, (short)0, acc[nt], false, false);
        }
        __syncthreads();
    }

    if (!active) return;

    #pragma unroll
    for (int nt = 0; nt < 8; ++nt) {
        const int col = nt * 16 + l15;
        const float bv = bias[col];
        #pragma unroll
        for (int j = 0; j < 8; ++j) {
            const int r = rowBase + j + half * 8;   // C layout: VGPR j -> M=j / j+8
            float v = acc[nt][j] + bv;
            if (RELU) v = fmaxf(v, 0.0f);
            if (OUT_F32)
                ((float*)outp)[(size_t)r * ldo + col] = v;
            else
                ((u16*)outp)[(size_t)r * ldo + col] = f2bf(v);
        }
    }
}

// ---------------------------------------------------------------------------
// E1: target-target edges. One wave per edge; lane handles 4 bf16 (2 pk ops).
// ---------------------------------------------------------------------------
__global__ __launch_bounds__(256) void edge_tt_kernel(
        const int* __restrict__ ett, u16* __restrict__ A_big,
        float* __restrict__ cnt_s2d, float* __restrict__ cnt_d2s, int nE) {
    const int wid  = (blockIdx.x * blockDim.x + threadIdx.x) >> 5;
    const int lane = threadIdx.x & 31;
    if (wid >= nE) return;
    __builtin_prefetch(ett + wid + 4096, 0, 1);           // global_prefetch

    const int src = ett[wid];
    const int dst = ett[nE + wid];

    const uint2 s = ((const uint2*)(A_big + (size_t)src * 512))[lane];
    const uint2 d = ((const uint2*)(A_big + (size_t)dst * 512))[lane];

    u16* p_s2d = A_big + (size_t)dst * 512 + 128 + lane * 4;  // x_t[src] -> bucket dst
    u16* p_d2s = A_big + (size_t)src * 512 + 256 + lane * 4;  // x_t[dst] -> bucket src
    pk_add_bf16(p_s2d,     s.x);
    pk_add_bf16(p_s2d + 2, s.y);
    pk_add_bf16(p_d2s,     d.x);
    pk_add_bf16(p_d2s + 2, d.y);

    if (lane == 0) {
        atomicAdd(cnt_s2d + dst, 1.0f);
        atomicAdd(cnt_d2s + src, 1.0f);
    }
}

// ---------------------------------------------------------------------------
// E2: context-target edges
// ---------------------------------------------------------------------------
__global__ __launch_bounds__(256) void edge_ct_kernel(
        const int* __restrict__ esrc, const int* __restrict__ edst,
        const u16* __restrict__ xc_h, u16* __restrict__ A_big,
        float* __restrict__ cnt_ct, int nE) {
    const int wid  = (blockIdx.x * blockDim.x + threadIdx.x) >> 5;
    const int lane = threadIdx.x & 31;
    if (wid >= nE) return;
    __builtin_prefetch(esrc + wid + 4096, 0, 1);

    const int src = esrc[wid];
    const int dst = edst[wid];

    const uint2 s = ((const uint2*)(xc_h + (size_t)src * 128))[lane];
    u16* p = A_big + (size_t)dst * 512 + 384 + lane * 4;
    pk_add_bf16(p,     s.x);
    pk_add_bf16(p + 2, s.y);
    if (lane == 0) atomicAdd(cnt_ct + dst, 1.0f);
}

// ---------------------------------------------------------------------------
// N: divide agg columns by max(count,1)
// ---------------------------------------------------------------------------
__global__ void normalize_agg(u16* __restrict__ A_big, const float* __restrict__ cnts) {
    const int total  = NT * 192;   // 384 agg cols / 2 per thread
    const int stride = gridDim.x * blockDim.x;
    for (int i = blockIdx.x * blockDim.x + threadIdx.x; i < total; i += stride) {
        const int row = i / 192;
        const int p   = i % 192;          // pair index within agg region
        const int seg = p >> 6;           // 0: s2d, 1: d2s, 2: ct
        const float cnt = cnts[seg * NT + row];
        const float inv = 1.0f / fmaxf(cnt, 1.0f);
        u32* ptr = (u32*)(A_big + (size_t)row * 512 + 128 + p * 2);
        u32 v = *ptr;
        float lo = bf2f((u16)(v & 0xFFFFu)) * inv;
        float hi = bf2f((u16)(v >> 16)) * inv;
        *ptr = (u32)f2bf(lo) | ((u32)f2bf(hi) << 16);
    }
}

// ---------------------------------------------------------------------------
// host side
// ---------------------------------------------------------------------------
static inline size_t align256(size_t x) { return (x + 255) & ~(size_t)255; }

extern "C" void kernel_launch(void* const* d_in, const int* in_sizes, int n_in,
                              void* d_out, int out_size, void* d_ws, size_t ws_size,
                              hipStream_t stream) {
    (void)in_sizes; (void)n_in; (void)out_size; (void)ws_size;

    const float* x_target  = (const float*)d_in[0];
    const float* x_context = (const float*)d_in[1];
    const int*   edge_tt   = (const int*)d_in[2];
    const int*   ect_src   = (const int*)d_in[3];
    const int*   ect_dst   = (const int*)d_in[4];
    const float* Wp_t   = (const float*)d_in[5];
    const float* bp_t   = (const float*)d_in[6];
    const float* Wp_c   = (const float*)d_in[7];
    const float* bp_c   = (const float*)d_in[8];
    const float* W_self = (const float*)d_in[9];
    const float* b_self = (const float*)d_in[10];
    const float* W_s2d  = (const float*)d_in[11];
    const float* b_s2d  = (const float*)d_in[12];
    const float* W_d2s  = (const float*)d_in[13];
    const float* b_d2s  = (const float*)d_in[14];
    const float* W_ct_l = (const float*)d_in[15];
    const float* b_ct_l = (const float*)d_in[16];
    const float* W_ct_r = (const float*)d_in[17];
    const float* W_out  = (const float*)d_in[18];
    const float* b_out  = (const float*)d_in[19];

    // workspace carve
    char* base = (char*)d_ws;
    size_t off = 0;
    u16* A_big  = (u16*)(base + off); off = align256(off + (size_t)NT * 512 * 2);
    u16* zr     = (u16*)(base + off); off = align256(off + (size_t)NT * 128 * 2);
    u16* xt_in  = (u16*)(base + off); off = align256(off + (size_t)NT * 128 * 2);
    u16* xc_in  = (u16*)(base + off); off = align256(off + (size_t)NC * 128 * 2);
    u16* xc_h   = (u16*)(base + off); off = align256(off + (size_t)NC * 128 * 2);
    float* cnts = (float*)(base + off); off = align256(off + (size_t)3 * NT * 4);
    u16* WpT_t  = (u16*)(base + off); off = align256(off + (size_t)H * H * 2);
    u16* WpT_c  = (u16*)(base + off); off = align256(off + (size_t)H * H * 2);
    u16* WcombT = (u16*)(base + off); off = align256(off + (size_t)H * 512 * 2);
    u16* WoutT  = (u16*)(base + off); off = align256(off + (size_t)H * H * 2);
    float* b_comb = (float*)(base + off); off = align256(off + (size_t)H * 4);

    // P0: weight prep
    prep_weights<<<128, 256, 0, stream>>>(Wp_t, Wp_c, W_self, W_ct_r, W_s2d, W_d2s,
                                          W_ct_l, W_out, b_self, b_s2d, b_d2s, b_ct_l,
                                          WpT_t, WpT_c, WcombT, WoutT, b_comb);
    // P1: input conversion
    conv_f32_bf16<<<2048, 256, 0, stream>>>(x_target, xt_in, NT * 128);
    conv_f32_bf16<<<512, 256, 0, stream>>>(x_context, xc_in, NC * 128);
    // P2: zero aggregation region + counters
    zero_agg<<<(NT * 48 + 255) / 256, 256, 0, stream>>>(A_big);
    zero_f32<<<(3 * NT + 255) / 256, 256, 0, stream>>>(cnts, 3 * NT);

    // K1/K2: per-type pretransform + relu
    const int blocksT = ((NT >> 4) + 7) / 8;   // 782
    const int blocksC = ((NC >> 4) + 7) / 8;   // 157
    gemm_bf16_n128<true, false><<<blocksT, 256, 0, stream>>>(
        xt_in, 128, WpT_t, 128, bp_t, (void*)A_big, 512, NT);
    gemm_bf16_n128<true, false><<<blocksC, 256, 0, stream>>>(
        xc_in, 128, WpT_c, 128, bp_c, (void*)xc_h, 128, NC);

    // E1/E2: scatter aggregation via packed bf16 atomics
    edge_tt_kernel<<<(ETT + 7) / 8, 256, 0, stream>>>(edge_tt, A_big, cnts, cnts + NT, ETT);
    edge_ct_kernel<<<(ECT + 7) / 8, 256, 0, stream>>>(ect_src, ect_dst, xc_h, A_big,
                                                      cnts + 2 * NT, ECT);
    // N: mean normalization
    normalize_agg<<<8192, 256, 0, stream>>>(A_big, cnts);

    // G1: fused K=512 GEMM (+identity residual folded) + relu
    gemm_bf16_n128<true, false><<<blocksT, 256, 0, stream>>>(
        A_big, 512, WcombT, 512, b_comb, (void*)zr, 128, NT);
    // G2: output projection, f32 out
    gemm_bf16_n128<false, true><<<blocksT, 256, 0, stream>>>(
        zr, 128, WoutT, 128, b_out, d_out, 128, NT);
}
